// EncoderWeightedICGCN_3917010174723
// MI455X (gfx1250) — matrix operand order repaired and checked by reading, not compiled, not run on hardware
//
// CDNA5 / gfx1250 implementation of EncoderWeightedICGCN forward.
//
// Strategy: hoist per-node encoders out of the per-edge loop (16x fewer FLOPs
// there), run all remaining dense math as f16 WMMA (v_wmma_f32_16x16x32_f16,
// f32 accumulate), f32 atomics for segment sums.
//
// Round-1 fix: libm tanhf expanded to a branchy ~40-op sequence with EXEC
// save/restore in the hot edge kernel; replaced with a branch-free
// exp-based tanh and single-instruction v_rcp_f32 sigmoid.
//
// NOTE: d_in flattening assumes setup_inputs() insertion order with the
// 'params' pytree flattened depth-first in insertion order.
// NOTE: float atomicAdd makes reduction order (last-bit rounding)
// nondeterministic; values are tolerance-stable.

#include <hip/hip_runtime.h>
#include <math.h>

typedef __attribute__((ext_vector_type(16))) _Float16 v16h;
typedef __attribute__((ext_vector_type(8)))  float    v8f;
typedef unsigned int u32;

static constexpr int N_STATE  = 50000;
static constexpr int N_ACTION = 50000;
static constexpr int N_EDGE   = 800000;

// ---------------------------------------------------------------- helpers ---

__device__ __forceinline__ void lds_fence() {
  // gfx1250 split counters: wait for all outstanding DS ops of this wave so
  // cross-lane LDS writes are visible before the next phase reads them.
  asm volatile("s_wait_dscnt 0" ::: "memory");
}

// Branch-free sigmoid: one v_exp_f32 + one v_rcp_f32, no EXEC divergence.
__device__ __forceinline__ float sigf(float x) {
  return __builtin_amdgcn_rcpf(1.f + __expf(-x));
}

// Branch-free tanh: clamp keeps exp finite, (e-1)*rcp(e+1) is exact at the
// f16 storage precision we feed the next WMMA layer with.
__device__ __forceinline__ float fast_tanhf(float x) {
  float xc = fminf(fmaxf(x, -15.f), 15.f);
  float e  = __expf(2.f * xc);
  return (e - 1.f) * __builtin_amdgcn_rcpf(e + 1.f);
}

__device__ __forceinline__ float softplusf(float x) {
  return (x > 20.f) ? x : log1pf(__expf(x));
}

__device__ __forceinline__ v8f wmma_f16(v16h a, v16h b, v8f c) {
  // (neg_a, A, neg_b, B, c_mod, C, reuse_a, reuse_b)
  return __builtin_amdgcn_wmma_f32_16x16x32_f16(false, a, false, b, (short)0, c,
                                                false, false);
}

// Load a 16x32 f16 A-fragment from an LDS tile (row-major, row stride `rs`
// halves, even). Layout per CDNA5 ISA 7.12.2 (16-bit A 16x32):
//   lanes 0-15 : row M=lane,    V0..V3 <- K 0..7,   V4..V7 <- K 16..23
//   lanes 16-31: row M=lane-16, V0..V3 <- K 8..15,  V4..V7 <- K 24..31
// Contiguous dword groups merge into two ds_load_b128 per fragment.
__device__ __forceinline__ v16h load_a_frag(const _Float16* T, int rs, int lane,
                                            int kbase) {
  const int row = lane & 15;
  const int ko  = (lane < 16) ? 0 : 8;
  const _Float16* r = T + row * rs + kbase;
  union { v16h v; u32 d[8]; } f;
#pragma unroll
  for (int j = 0; j < 4; ++j) f.d[j]     = *(const u32*)(r + ko + 2 * j);
#pragma unroll
  for (int j = 0; j < 4; ++j) f.d[4 + j] = *(const u32*)(r + 16 + ko + 2 * j);
  return f.v;
}

// B-fragments are pre-swizzled in global memory: 256 dwords per (ktile,ntile),
// lane-major (lane*8 + j). Two global_load_b128 per fragment, L2-resident.
__device__ __forceinline__ v16h load_b_frag(const u32* __restrict__ frag, int lane) {
  union { v16h v; uint4 q[2]; } f;
  const uint4* p = (const uint4*)frag + lane * 2;
  f.q[0] = p[0];
  f.q[1] = p[1];
  return f.v;
}

// Activate a D accumulator (16x16 f32) and store to an LDS f16 tile.
// D layout: lanes 0-15: N=lane, VGPR r -> M=r; lanes 16-31: N=lane-16, M=8+r.
// mode: 0=tanh, 1=sigmoid, 2=cprelu(slope), 3=identity
__device__ __forceinline__ void store_d_act(_Float16* T, int rs, int lane, int nt,
                                            v8f c, float bias, int mode, float slope) {
  const int n  = nt * 16 + (lane & 15);
  const int mb = (lane < 16) ? 0 : 8;
#pragma unroll
  for (int r = 0; r < 8; ++r) {
    float z = c[r] + bias;
    float v;
    if      (mode == 0) v = fast_tanhf(z);
    else if (mode == 1) v = sigf(z);
    else if (mode == 2) v = (z > 0.f) ? z : slope * z;
    else                v = z;
    T[(mb + r) * rs + n] = (_Float16)v;
  }
}

// ------------------------------------------------------------ prep kernels ---

__global__ void zero_f32_kernel(float* __restrict__ p, int n) {
  int i = blockIdx.x * blockDim.x + threadIdx.x;
  int s = gridDim.x * blockDim.x;
  for (; i < n; i += s) p[i] = 0.f;
}

__global__ void softplus_copy_kernel(const float* __restrict__ in,
                                     float* __restrict__ out, int n) {
  int i = blockIdx.x * blockDim.x + threadIdx.x;
  if (i < n) out[i] = softplusf(in[i]);
}

__global__ void degree_kernel(const int* __restrict__ dst, float* __restrict__ cnt,
                              int n) {
  int i = blockIdx.x * blockDim.x + threadIdx.x;
  if (i < n) atomicAdd(&cnt[dst[i]], 1.0f);
}

// Pre-swizzle a row-major weight W[64][K] (out,in) into WMMA B-fragments for
// B = W^T (KxN). B layout (per CDNA5 sparse-B pattern, dense 16-bit 32x16):
//   lanes 0-15 : col N=lane%16,  V0..V3 <- K 0..7,  V4..V7 <- K 16..23
//   lanes 16-31: col N=lane%16,  V0..V3 <- K 8..15, V4..V7 <- K 24..31
// Optionally applies softplus (NonnegativeLinear weight constraint).
__global__ void prep_frags_kernel(const float* __restrict__ W, u32* __restrict__ out,
                                  int K, int ktiles, int use_sp) {
  int idx = blockIdx.x * blockDim.x + threadIdx.x;
  int total = ktiles * 1024;               // 4 ntiles * 256 dwords per ktile
  if (idx >= total) return;
  int j  = idx & 7;
  int l  = (idx >> 3) & 31;
  int nt = (idx >> 8) & 3;
  int kt = idx >> 10;
  int n  = nt * 16 + (l & 15);
  int k0 = kt * 32 + ((l < 16) ? 0 : 8) + 2 * j + ((j >= 4) ? 8 : 0);
  float a = (k0     < K) ? W[n * K + k0]     : 0.f;
  float b = (k0 + 1 < K) ? W[n * K + k0 + 1] : 0.f;
  if (use_sp) { a = softplusf(a); b = softplusf(b); }
  _Float16 ha = (_Float16)a, hb = (_Float16)b;
  unsigned short ua = *(unsigned short*)&ha, ub = *(unsigned short*)&hb;
  out[idx] = (u32)ua | ((u32)ub << 16);
}

// ------------------------------------------------- per-node encoder (VALU) ---
// encU = _enc_u(u) [NA,64] / encH = _enc_h(h) [NS,64]; ~1.5 GFLOP total, so
// plain VALU is fine. One wave per node; weights already softplus'd.

__global__ __launch_bounds__(128) void encode_nodes_kernel(
    int nNodes, int inDim, const float* __restrict__ X,
    const float* __restrict__ w1, const float* __restrict__ b1,
    const float* __restrict__ a1,
    const float* __restrict__ w2, const float* __restrict__ b2,
    const float* __restrict__ a2,
    const float* __restrict__ w3, const float* __restrict__ b3,
    int hasBias, float* __restrict__ out) {
  __shared__ float sZ[4][2][64];
  const int wave = threadIdx.x >> 5;
  const int lane = threadIdx.x & 31;
  const int node = blockIdx.x * 4 + wave;
  if (node >= nNodes) return;
  float (*Z)[64] = sZ[wave];
  const float* x = X + (size_t)node * inDim;

#pragma unroll
  for (int hh = 0; hh < 2; ++hh) {
    const int o = lane + 32 * hh;
    float acc = hasBias ? b1[o] : 0.f;
    for (int k = 0; k < inDim; ++k) acc = fmaf(w1[o * inDim + k], x[k], acc);
    const float sl = sigf(a1[o]);
    Z[0][o] = (acc > 0.f) ? acc : sl * acc;
  }
  lds_fence();
#pragma unroll
  for (int hh = 0; hh < 2; ++hh) {
    const int o = lane + 32 * hh;
    float acc = hasBias ? b2[o] : 0.f;
    for (int k = 0; k < 64; ++k) acc = fmaf(w2[o * 64 + k], Z[0][k], acc);
    const float sl = sigf(a2[o]);
    Z[1][o] = (acc > 0.f) ? acc : sl * acc;
  }
  lds_fence();
#pragma unroll
  for (int hh = 0; hh < 2; ++hh) {
    const int o = lane + 32 * hh;
    float acc = hasBias ? b3[o] : 0.f;
    for (int k = 0; k < 64; ++k) acc = fmaf(w3[o * 64 + k], Z[1][k], acc);
    out[(size_t)node * 64 + o] = acc;
  }
}

// ----------------------------------------- edge gate MLP + gather/scatter ---
// One wave per 16-edge tile. Gate MLP 5->64->64->64 (tanh,tanh,sigmoid) done
// with WMMA f16 (input padded 5->32). Then msg = gate * enc[src] gathered per
// edge, atomically accumulated into accum[dst].

__global__ __launch_bounds__(128) void edge_mlp_kernel(
    int ntiles,
    const int* __restrict__ src, const int* __restrict__ dst,
    const float* __restrict__ posSrc, const float* __restrict__ posDst,
    const float* __restrict__ dis,
    const float* __restrict__ enc,
    const float* __restrict__ b1, const float* __restrict__ b2,
    const float* __restrict__ b3,
    const u32* __restrict__ f1, const u32* __restrict__ f2,
    const u32* __restrict__ f3,
    float* __restrict__ accum) {
  __shared__ _Float16 sIN[4][16 * 40];   // 16x32 padded input (+8/row pad)
  __shared__ _Float16 sACT[4][16 * 72];  // 16x64 activations  (+8/row pad)
  const int wave = threadIdx.x >> 5;
  const int lane = threadIdx.x & 31;
  const int tile = blockIdx.x * 4 + wave;
  if (tile >= ntiles) return;
  _Float16* IN  = sIN[wave];
  _Float16* ACT = sACT[wave];
  const int e0  = tile * 16;
  const int n15 = lane & 15;

  for (int i = lane; i < 16 * 40 / 2; i += 32) ((u32*)IN)[i] = 0u;
  lds_fence();
  if (lane < 16) {
    const int e = e0 + lane;
    const int s = src[e], d = dst[e];
    _Float16* r = IN + lane * 40;
    r[0] = (_Float16)posSrc[2 * s + 0];
    r[1] = (_Float16)posSrc[2 * s + 1];
    r[2] = (_Float16)posDst[2 * d + 0];
    r[3] = (_Float16)posDst[2 * d + 1];
    r[4] = (_Float16)dis[e];
  }
  lds_fence();

  { // layer 1: [16,32] x [32,64], tanh
    v16h a = load_a_frag(IN, 40, lane, 0);
#pragma unroll
    for (int nt = 0; nt < 4; ++nt) {
      v8f c = {};
      c = wmma_f16(a, load_b_frag(f1 + nt * 256, lane), c);
      store_d_act(ACT, 72, lane, nt, c, b1[nt * 16 + n15], 0, 0.f);
    }
  }
  lds_fence();
  { // layer 2: [16,64] x [64,64], tanh
    v16h a0 = load_a_frag(ACT, 72, lane, 0);
    v16h a1 = load_a_frag(ACT, 72, lane, 32);
#pragma unroll
    for (int nt = 0; nt < 4; ++nt) {
      v8f c = {};
      c = wmma_f16(a0, load_b_frag(f2 + (0 * 4 + nt) * 256, lane), c);
      c = wmma_f16(a1, load_b_frag(f2 + (1 * 4 + nt) * 256, lane), c);
      store_d_act(ACT, 72, lane, nt, c, b2[nt * 16 + n15], 0, 0.f);
    }
  }
  lds_fence();
  { // layer 3: [16,64] x [64,64], sigmoid -> gate
    v16h a0 = load_a_frag(ACT, 72, lane, 0);
    v16h a1 = load_a_frag(ACT, 72, lane, 32);
#pragma unroll
    for (int nt = 0; nt < 4; ++nt) {
      v8f c = {};
      c = wmma_f16(a0, load_b_frag(f3 + (0 * 4 + nt) * 256, lane), c);
      c = wmma_f16(a1, load_b_frag(f3 + (1 * 4 + nt) * 256, lane), c);
      store_d_act(ACT, 72, lane, nt, c, b3[nt * 16 + n15], 1, 0.f);
    }
  }
  lds_fence();

  // msg = gate * enc[src]; atomic segment-sum into accum[dst]
  for (int m = 0; m < 16; ++m) {
    const int e = e0 + m;
    const int s = src[e];
    const int d = dst[e];
    if (m + 1 < 16) __builtin_prefetch(enc + (size_t)src[e + 1] * 64, 0, 0);
    const float g0 = (float)ACT[m * 72 + lane];
    const float g1 = (float)ACT[m * 72 + 32 + lane];
    const float* ev = enc + (size_t)s * 64;
    atomicAdd(&accum[(size_t)d * 64 + lane],      g0 * ev[lane]);
    atomicAdd(&accum[(size_t)d * 64 + 32 + lane], g1 * ev[32 + lane]);
  }
}

// --------------------------------------------------- PICNN node update ------
// One wave per 16-node tile. y=[h|sum_u|mean_h] (192, 6 ktiles); x=pos padded
// to one 32-ktile so x@wx.T / x@wxy.T also run through WMMA chains.

__global__ __launch_bounds__(128) void node_update_kernel(
    int ntiles,
    const float* __restrict__ pos, const float* __restrict__ hin,
    const float* __restrict__ sum_u, const float* __restrict__ sum_h,
    const float* __restrict__ cnt,
    const float* __restrict__ bx0, const float* __restrict__ by0,
    const float* __restrict__ sl0,
    const float* __restrict__ bx1, const float* __restrict__ by1,
    const float* __restrict__ sl1,
    const float* __restrict__ by2,
    const u32* __restrict__ fwy0, const u32* __restrict__ fwxy0,
    const u32* __restrict__ fwx0,
    const u32* __restrict__ fwy1, const u32* __restrict__ fwxy1,
    const u32* __restrict__ fwx1,
    const u32* __restrict__ fwy2, const u32* __restrict__ fwxy2,
    float* __restrict__ out) {
  __shared__ _Float16 sY[4][16 * 200];   // y tile 16x192 (+8 pad)
  __shared__ _Float16 sX[4][16 * 40];    // x tile 16x32 (pos in cols 0..1)
  __shared__ _Float16 sXA[4][16 * 72];   // x-path activations 16x64
  const int wave = threadIdx.x >> 5;
  const int lane = threadIdx.x & 31;
  const int tile = blockIdx.x * 4 + wave;
  if (tile >= ntiles) return;
  _Float16* Y  = sY[wave];
  _Float16* X  = sX[wave];
  _Float16* XA = sXA[wave];
  const int n0  = tile * 16;
  const int n15 = lane & 15;

  for (int i = lane; i < 16 * 40 / 2; i += 32) ((u32*)X)[i] = 0u;
  lds_fence();
  if (lane < 16) {
    const int nd = n0 + lane;
    const float c   = cnt[nd];
    const float inv = (c > 0.f) ? __builtin_amdgcn_rcpf(fmaxf(c, 1.f)) : 0.f;
    _Float16* r = Y + lane * 200;
    const float* hp = hin   + (size_t)nd * 64;
    const float* up = sum_u + (size_t)nd * 64;
    const float* mp = sum_h + (size_t)nd * 64;
    for (int k = 0; k < 64; ++k) {
      r[k]       = (_Float16)hp[k];
      r[64 + k]  = (_Float16)up[k];
      r[128 + k] = (_Float16)(mp[k] * inv);
    }
    _Float16* rx = X + lane * 40;
    rx[0] = (_Float16)pos[2 * nd + 0];
    rx[1] = (_Float16)pos[2 * nd + 1];
  }
  lds_fence();

  { // layer 0
    v16h ax = load_a_frag(X, 40, lane, 0);
    v16h ay[6];
#pragma unroll
    for (int kt = 0; kt < 6; ++kt) ay[kt] = load_a_frag(Y, 200, lane, kt * 32);
#pragma unroll
    for (int nt = 0; nt < 4; ++nt) { // nx0 = tanh(x@wx0^T + bx0) -> XA
      v8f c = {};
      c = wmma_f16(ax, load_b_frag(fwx0 + nt * 256, lane), c);
      store_d_act(XA, 72, lane, nt, c, bx0[nt * 16 + n15], 0, 0.f);
    }
#pragma unroll
    for (int nt = 0; nt < 4; ++nt) { // ny0 = cprelu(y@sp(wy0)^T + x@wxy0^T + by0) -> Y
      v8f c = {};
#pragma unroll
      for (int kt = 0; kt < 6; ++kt)
        c = wmma_f16(ay[kt], load_b_frag(fwy0 + (kt * 4 + nt) * 256, lane), c);
      c = wmma_f16(ax, load_b_frag(fwxy0 + nt * 256, lane), c);
      store_d_act(Y, 200, lane, nt, c, by0[nt * 16 + n15], 2,
                  sigf(sl0[nt * 16 + n15]));
    }
  }
  lds_fence();
  { // layer 1
    v16h ay0 = load_a_frag(Y, 200, lane, 0);
    v16h ay1 = load_a_frag(Y, 200, lane, 32);
    v16h ax0 = load_a_frag(XA, 72, lane, 0);
    v16h ax1 = load_a_frag(XA, 72, lane, 32);
#pragma unroll
    for (int nt = 0; nt < 4; ++nt) { // nx1 = tanh(nx0@wx1^T + bx1) -> XA
      v8f c = {};
      c = wmma_f16(ax0, load_b_frag(fwx1 + (0 * 4 + nt) * 256, lane), c);
      c = wmma_f16(ax1, load_b_frag(fwx1 + (1 * 4 + nt) * 256, lane), c);
      store_d_act(XA, 72, lane, nt, c, bx1[nt * 16 + n15], 0, 0.f);
    }
#pragma unroll
    for (int nt = 0; nt < 4; ++nt) { // ny1 = cprelu(ny0@sp(wy1)^T + nx0@wxy1^T + by1) -> Y
      v8f c = {};
      c = wmma_f16(ay0, load_b_frag(fwy1 + (0 * 4 + nt) * 256, lane), c);
      c = wmma_f16(ay1, load_b_frag(fwy1 + (1 * 4 + nt) * 256, lane), c);
      c = wmma_f16(ax0, load_b_frag(fwxy1 + (0 * 4 + nt) * 256, lane), c);
      c = wmma_f16(ax1, load_b_frag(fwxy1 + (1 * 4 + nt) * 256, lane), c);
      store_d_act(Y, 200, lane, nt, c, by1[nt * 16 + n15], 2,
                  sigf(sl1[nt * 16 + n15]));
    }
  }
  lds_fence();
  { // layer 2: ny2 = ny1@sp(wy2)^T + nx1@wxy2^T + by2 -> out (f32)
    v16h ay0 = load_a_frag(Y, 200, lane, 0);
    v16h ay1 = load_a_frag(Y, 200, lane, 32);
    v16h ax0 = load_a_frag(XA, 72, lane, 0);
    v16h ax1 = load_a_frag(XA, 72, lane, 32);
#pragma unroll
    for (int nt = 0; nt < 4; ++nt) {
      v8f c = {};
      c = wmma_f16(ay0, load_b_frag(fwy2 + (0 * 4 + nt) * 256, lane), c);
      c = wmma_f16(ay1, load_b_frag(fwy2 + (1 * 4 + nt) * 256, lane), c);
      c = wmma_f16(ax0, load_b_frag(fwxy2 + (0 * 4 + nt) * 256, lane), c);
      c = wmma_f16(ax1, load_b_frag(fwxy2 + (1 * 4 + nt) * 256, lane), c);
      const int n = nt * 16 + n15;
      const float b  = by2[n];
      const int   mb = (lane < 16) ? 0 : 8;
#pragma unroll
      for (int r = 0; r < 8; ++r)
        out[(size_t)(n0 + mb + r) * 64 + n] = c[r] + b;
    }
  }
}

// ------------------------------------------------------------------ host ----

extern "C" void kernel_launch(void* const* d_in, const int* in_sizes, int n_in,
                              void* d_out, int out_size, void* d_ws, size_t ws_size,
                              hipStream_t stream) {
  // setup_inputs() insertion order; 'params' flattened depth-first:
  const float* pos_state  = (const float*)d_in[0];
  const float* pos_action = (const float*)d_in[1];
  const float* h_in       = (const float*)d_in[2];
  const float* u_in       = (const float*)d_in[3];
  const float* dis_a2s    = (const float*)d_in[4];
  const float* dis_s2s    = (const float*)d_in[5];
  const int*   a2s_src    = (const int*)d_in[6];
  const int*   a2s_dst    = (const int*)d_in[7];
  const int*   s2s_src    = (const int*)d_in[8];
  const int*   s2s_dst    = (const int*)d_in[9];
  int p = 10;
  const float* ud_w1 = (const float*)d_in[p++]; // u2h_dis: w1[64,5]
  const float* ud_b1 = (const float*)d_in[p++];
  const float* ud_w2 = (const float*)d_in[p++]; // [64,64]
  const float* ud_b2 = (const float*)d_in[p++];
  const float* ud_w3 = (const float*)d_in[p++]; // [64,64]
  const float* ud_b3 = (const float*)d_in[p++];
  const float* hd_w1 = (const float*)d_in[p++]; // h2h_dis
  const float* hd_b1 = (const float*)d_in[p++];
  const float* hd_w2 = (const float*)d_in[p++];
  const float* hd_b2 = (const float*)d_in[p++];
  const float* hd_w3 = (const float*)d_in[p++];
  const float* hd_b3 = (const float*)d_in[p++];
  const float* uu_w1 = (const float*)d_in[p++]; // u2h_u: w1[64,8]
  const float* uu_a1 = (const float*)d_in[p++];
  const float* uu_w2 = (const float*)d_in[p++];
  const float* uu_a2 = (const float*)d_in[p++];
  const float* uu_w3 = (const float*)d_in[p++];
  const float* hh_w1 = (const float*)d_in[p++]; // h2h_h: w1[64,64]
  const float* hh_b1 = (const float*)d_in[p++];
  const float* hh_a1 = (const float*)d_in[p++];
  const float* hh_w2 = (const float*)d_in[p++];
  const float* hh_b2 = (const float*)d_in[p++];
  const float* hh_a2 = (const float*)d_in[p++];
  const float* hh_w3 = (const float*)d_in[p++];
  const float* hh_b3 = (const float*)d_in[p++];
  const float* u0_wx  = (const float*)d_in[p++]; // upd[0]: wx[64,2]
  const float* u0_bx  = (const float*)d_in[p++];
  const float* u0_wy  = (const float*)d_in[p++]; // [64,192]
  const float* u0_by  = (const float*)d_in[p++];
  const float* u0_wxy = (const float*)d_in[p++]; // [64,2]
  const float* u0_a   = (const float*)d_in[p++];
  const float* u1_wx  = (const float*)d_in[p++]; // upd[1]: [64,64] each
  const float* u1_bx  = (const float*)d_in[p++];
  const float* u1_wy  = (const float*)d_in[p++];
  const float* u1_by  = (const float*)d_in[p++];
  const float* u1_wxy = (const float*)d_in[p++];
  const float* u1_a   = (const float*)d_in[p++];
  const float* u2_wx  = (const float*)d_in[p++]; // upd[2]: wx[1,64] (unused)
  const float* u2_bx  = (const float*)d_in[p++]; // (unused)
  const float* u2_wy  = (const float*)d_in[p++]; // [64,64]
  const float* u2_by  = (const float*)d_in[p++];
  const float* u2_wxy = (const float*)d_in[p++]; // [64,64]
  (void)u2_wx; (void)u2_bx; (void)n_in; (void)in_sizes; (void)out_size; (void)ws_size;

  // ---- workspace carve (floats) ----
  float* ws = (float*)d_ws;
  size_t off = 0;
  auto alloc = [&](size_t n) { float* r = ws + off; off += n; return r; };
  float* encU  = alloc((size_t)N_ACTION * 64);
  float* encH  = alloc((size_t)N_STATE * 64);
  float* sum_u = alloc((size_t)N_STATE * 64);  // sum_u, sum_h, cnt contiguous
  float* sum_h = alloc((size_t)N_STATE * 64);
  float* cntv  = alloc((size_t)N_STATE);
  float* sp_uw1 = alloc(64 * 8);
  float* sp_uw2 = alloc(64 * 64);
  float* sp_uw3 = alloc(64 * 64);
  float* sp_hw1 = alloc(64 * 64);
  float* sp_hw2 = alloc(64 * 64);
  float* sp_hw3 = alloc(64 * 64);
  u32* frags = (u32*)(ws + off);
  size_t foff = 0;
  auto falloc = [&](int ktiles) { u32* r = frags + foff; foff += (size_t)ktiles * 1024; return r; };
  u32* fA1  = falloc(1); u32* fA2  = falloc(2); u32* fA3  = falloc(2); // a2s gate
  u32* fS1  = falloc(1); u32* fS2  = falloc(2); u32* fS3  = falloc(2); // s2s gate
  u32* fWY0 = falloc(6); u32* fWXY0 = falloc(1); u32* fWX0 = falloc(1);
  u32* fWY1 = falloc(2); u32* fWXY1 = falloc(2); u32* fWX1 = falloc(2);
  u32* fWY2 = falloc(2); u32* fWXY2 = falloc(2);

  // ---- 1. zero accumulators ----
  zero_f32_kernel<<<2048, 256, 0, stream>>>(sum_u, N_STATE * 64 * 2 + N_STATE);

  // ---- 2. softplus'd encoder weights ----
  auto spc = [&](const float* a, float* b, int n) {
    softplus_copy_kernel<<<(n + 255) / 256, 256, 0, stream>>>(a, b, n);
  };
  spc(uu_w1, sp_uw1, 64 * 8);
  spc(uu_w2, sp_uw2, 64 * 64);
  spc(uu_w3, sp_uw3, 64 * 64);
  spc(hh_w1, sp_hw1, 64 * 64);
  spc(hh_w2, sp_hw2, 64 * 64);
  spc(hh_w3, sp_hw3, 64 * 64);

  // ---- 3. pre-swizzle WMMA B-fragments ----
  auto mkfr = [&](const float* W, u32* dstF, int K, int kt, int sp) {
    int total = kt * 1024;
    prep_frags_kernel<<<(total + 255) / 256, 256, 0, stream>>>(W, dstF, K, kt, sp);
  };
  mkfr(ud_w1, fA1, 5, 1, 0);   mkfr(ud_w2, fA2, 64, 2, 0);  mkfr(ud_w3, fA3, 64, 2, 0);
  mkfr(hd_w1, fS1, 5, 1, 0);   mkfr(hd_w2, fS2, 64, 2, 0);  mkfr(hd_w3, fS3, 64, 2, 0);
  mkfr(u0_wy, fWY0, 192, 6, 1); mkfr(u0_wxy, fWXY0, 2, 1, 0); mkfr(u0_wx, fWX0, 2, 1, 0);
  mkfr(u1_wy, fWY1, 64, 2, 1);  mkfr(u1_wxy, fWXY1, 64, 2, 0); mkfr(u1_wx, fWX1, 64, 2, 0);
  mkfr(u2_wy, fWY2, 64, 2, 1);  mkfr(u2_wxy, fWXY2, 64, 2, 0);

  // ---- 4. per-node encoders (hoisted out of edge loop: 16x fewer FLOPs) ----
  encode_nodes_kernel<<<(N_ACTION + 3) / 4, 128, 0, stream>>>(
      N_ACTION, 8, u_in, sp_uw1, uu_a1, uu_a1, sp_uw2, uu_a2, uu_a2,
      sp_uw3, uu_a2, /*hasBias=*/0, encU);
  encode_nodes_kernel<<<(N_STATE + 3) / 4, 128, 0, stream>>>(
      N_STATE, 64, h_in, sp_hw1, hh_b1, hh_a1, sp_hw2, hh_b2, hh_a2,
      sp_hw3, hh_b3, /*hasBias=*/1, encH);

  // ---- 5. degree for mean reduction (s2s) ----
  degree_kernel<<<(N_EDGE + 255) / 256, 256, 0, stream>>>(s2s_dst, cntv, N_EDGE);

  // ---- 6. edge passes (WMMA gate MLP + gather * enc + atomic scatter) ----
  const int etiles = N_EDGE / 16;
  edge_mlp_kernel<<<(etiles + 3) / 4, 128, 0, stream>>>(
      etiles, a2s_src, a2s_dst, pos_action, pos_state, dis_a2s, encU,
      ud_b1, ud_b2, ud_b3, fA1, fA2, fA3, sum_u);
  edge_mlp_kernel<<<(etiles + 3) / 4, 128, 0, stream>>>(
      etiles, s2s_src, s2s_dst, pos_state, pos_state, dis_s2s, encH,
      hd_b1, hd_b2, hd_b3, fS1, fS2, fS3, sum_h);

  // ---- 7. PICNN node update (all-WMMA) ----
  const int ntiles = N_STATE / 16;
  node_update_kernel<<<(ntiles + 3) / 4, 128, 0, stream>>>(
      ntiles, pos_state, h_in, sum_u, sum_h, cntv,
      u0_bx, u0_by, u0_a, u1_bx, u1_by, u1_a, u2_by,
      fWY0, fWXY0, fWX0, fWY1, fWXY1, fWX1, fWY2, fWXY2,
      (float*)d_out);
}
